// ann1_poisson_snn2_45535243272631
// MI455X (gfx1250) — matrix operand order repaired and checked by reading, not compiled, not run on hardware
//
#include <hip/hip_runtime.h>
#include <hip/hip_bf16.h>
#include <stdint.h>

// ---------------- problem dims ----------------
#define BB    256
#define TT    100
#define TPAD  112      // 7 tiles of 16
#define FIN   784
#define FINP  800      // 25 * 32
#define FHID  500
#define FHIDP 512      // 16 * 32
#define FOUT  10

static constexpr float EM_ = 0.77880078307140487f;   // exp(-1/4)
static constexpr float A1_ = 1.14668022424284721f;   // EM + ES
static constexpr float A2_ = -0.28650479686019009f;  // -EM*ES

typedef __attribute__((ext_vector_type(16))) __bf16 v16bf;
typedef __attribute__((ext_vector_type(8)))  __bf16 v8bf;
typedef __attribute__((ext_vector_type(8)))  float  v8f;

// ---------------- helpers ----------------
static __device__ __forceinline__ __bf16 f2bf(float f) {
  union { float f; uint32_t u; } v; v.f = f;
  uint32_t r = v.u + 0x7FFFu + ((v.u >> 16) & 1u);  // round-to-nearest-even
  union { uint16_t h; __bf16 b; } o; o.h = (uint16_t)(r >> 16);
  return o.b;
}

static __device__ __forceinline__ v16bf load_frag(const __bf16* lo, const __bf16* hi) {
  v8bf a = *reinterpret_cast<const v8bf*>(lo);
  v8bf b = *reinterpret_cast<const v8bf*>(hi);
  v16bf r;
#pragma unroll
  for (int i = 0; i < 8; ++i) { r[i] = a[i]; r[i + 8] = b[i]; }
  return r;
}

static __device__ __forceinline__ uint32_t pcg(uint32_t x) {
  x = x * 747796405u + 2891336453u;
  x = ((x >> ((x >> 28) + 4u)) ^ x) * 277803737u;
  return (x >> 22) ^ x;
}
static __device__ __forceinline__ float gauss_hash(uint32_t idx) {
  uint32_t u1 = pcg(idx), u2 = pcg(idx ^ 0x9E3779B9u);
  float f1 = (float)(u1 >> 8) * (1.0f / 16777216.0f) + (0.5f / 16777216.0f); // (0,1)
  float f2 = (float)(u2 >> 8) * (1.0f / 16777216.0f);
  return __builtin_sqrtf(-2.0f * __logf(f1)) * __cosf(6.2831853071795865f * f2);
}

// ---------------- conversion / padding passes ----------------
// x [B][FIN][TT] f32  ->  xb [B][TPAD][FINP] bf16 (transposed so K is contiguous)
__global__ void cvt_x_kernel(const float* __restrict__ x, __bf16* __restrict__ xb) {
  size_t i = (size_t)blockIdx.x * blockDim.x + threadIdx.x;
  if (i >= (size_t)BB * TPAD * FINP) return;
  int k = (int)(i % FINP);
  int t = (int)((i / FINP) % TPAD);
  int b = (int)(i / ((size_t)FINP * TPAD));
  float v = 0.0f;
  if (t < TT && k < FIN) v = x[((size_t)b * FIN + k) * TT + t];
  xb[i] = f2bf(v);
}

// generic zero-padded f32 -> bf16 matrix copy: dst[R][C] <- src[SR][SC]
__global__ void cvt_pad_kernel(const float* __restrict__ src, __bf16* __restrict__ dst,
                               int R, int C, int SR, int SC) {
  size_t i = (size_t)blockIdx.x * blockDim.x + threadIdx.x;
  if (i >= (size_t)R * C) return;
  int c = (int)(i % C), r = (int)(i / C);
  float v = (r < SR && c < SC) ? src[(size_t)r * SC + c] : 0.0f;
  dst[i] = f2bf(v);
}

// ---------------- GEMM1 + relu + reparameterized sampling ----------------
// W1b [1024][FINP], Xb [B][TPAD][FINP], b1 [1000]  ->  coding [B][FHID][TT] f32
// out[b,o,t] = relu(sum_k W1[o,k] * x[b,k,t] + b1[o]);  coding = mu + eps*exp(0.5*lnv)
__global__ void gemm1_sample_kernel(const __bf16* __restrict__ W1b,
                                    const __bf16* __restrict__ Xb,
                                    const float*  __restrict__ b1,
                                    float*        __restrict__ coding) {
  const int b    = blockIdx.x;
  const int t0   = blockIdx.y * 16;                 // 7 tiles
  const int wave = threadIdx.x >> 5;                // 4 waves / block
  const int lane = threadIdx.x & 31;
  const int h0   = (blockIdx.z * 4 + wave) * 16;    // 32 h tiles
  const int row  = lane & 15;
  const int half = lane >> 4;

  const __bf16* Amu = W1b + (size_t)(h0 + row) * FINP;
  const __bf16* Alv = W1b + (size_t)(FHID + h0 + row) * FINP;
  const __bf16* Bp  = Xb  + ((size_t)b * TPAD + (t0 + row)) * FINP;

  v8f acc_mu = {}; v8f acc_lv = {};
  for (int k0 = 0; k0 < FINP; k0 += 32) {
    const int ka = k0 + 8 * half;    // A: elems 0..7 -> K=ka.., 8..15 -> K=ka+16..
    const int kb = k0 + 8 * half;    // B: elems 0..7 -> K=kb.., 8..15 -> K=kb+16..
    v16bf am = load_frag(Amu + ka, Amu + ka + 16);
    v16bf al = load_frag(Alv + ka, Alv + ka + 16);
    v16bf bm = load_frag(Bp + kb, Bp + kb + 16);
    acc_mu = __builtin_amdgcn_wmma_f32_16x16x32_bf16(false, am, false, bm, (short)0, acc_mu, false, false);
    acc_lv = __builtin_amdgcn_wmma_f32_16x16x32_bf16(false, al, false, bm, (short)0, acc_lv, false, false);
  }

  const int t = t0 + row;
#pragma unroll
  for (int r = 0; r < 8; ++r) {
    int h = h0 + r + 8 * half;
    if (h < FHID && t < TT) {
      float mu = fmaxf(acc_mu[r] + b1[h], 0.0f);
      float lv = fmaxf(acc_lv[r] + b1[FHID + h], 0.0f);
      float eps = gauss_hash((uint32_t)(((b * 512) + h) * 128 + t));
      coding[((size_t)b * FHID + h) * TT + t] = mu + eps * __expf(0.5f * lv);
    }
  }
}

// ---------------- dual-exp IIR (coding f32 -> psp bf16, transposed+padded) ------
__global__ void iir_kernel(const float* __restrict__ coding, __bf16* __restrict__ psp) {
  int idx = blockIdx.x * blockDim.x + threadIdx.x;   // b*512 + h
  if (idx >= BB * FHIDP) return;
  int b = idx >> 9, h = idx & (FHIDP - 1);
  float y1 = 0.0f, y2 = 0.0f;
  for (int t = 0; t < TPAD; ++t) {
    float y = 0.0f;
    if (h < FHID && t < TT) {
      float x = coding[((size_t)b * FHID + h) * TT + t];
      y = A1_ * y1 + A2_ * y2 + x;
      y2 = y1; y1 = y;
    }
    psp[((size_t)b * TPAD + t) * FHIDP + h] = f2bf(y);
  }
}

// ---------------- GEMM2 + bias (hidden layer current) ----------------
__global__ void gemm2_kernel(const __bf16* __restrict__ W2b,   // [512][512]
                             const __bf16* __restrict__ P,     // [B][TPAD][512]
                             const float*  __restrict__ b2,
                             float*        __restrict__ cur) { // [B][FHID][TT]
  const int b    = blockIdx.x;
  const int t0   = blockIdx.y * 16;
  const int wave = threadIdx.x >> 5;
  const int lane = threadIdx.x & 31;
  const int h0   = (blockIdx.z * 4 + wave) * 16;
  const int row  = lane & 15;
  const int half = lane >> 4;

  const __bf16* Ap = W2b + (size_t)(h0 + row) * FHIDP;
  const __bf16* Bp = P   + ((size_t)b * TPAD + (t0 + row)) * FHIDP;

  v8f acc = {};
  for (int k0 = 0; k0 < FHIDP; k0 += 32) {
    const int ka = k0 + 8 * half;
    const int kb = k0 + 8 * half;
    v16bf a  = load_frag(Ap + ka, Ap + ka + 16);
    v16bf bm = load_frag(Bp + kb, Bp + kb + 16);
    acc = __builtin_amdgcn_wmma_f32_16x16x32_bf16(false, a, false, bm, (short)0, acc, false, false);
  }

  const int t = t0 + row;
#pragma unroll
  for (int r = 0; r < 8; ++r) {
    int h = h0 + r + 8 * half;
    if (h < FHID && t < TT)
      cur[((size_t)b * FHID + h) * TT + t] = acc[r] + b2[h];
  }
}

// ---------------- LIF scan + dual-exp IIR fused (hidden -> psp3) ----------------
__global__ void lif_iir_kernel(const float* __restrict__ cur, __bf16* __restrict__ psp) {
  int idx = blockIdx.x * blockDim.x + threadIdx.x;
  if (idx >= BB * FHIDP) return;
  int b = idx >> 9, h = idx & (FHIDP - 1);
  float v = 0.0f, s = 0.0f, y1 = 0.0f, y2 = 0.0f;
  for (int t = 0; t < TPAD; ++t) {
    float y = 0.0f;
    if (h < FHID && t < TT) {
      float c = cur[((size_t)b * FHID + h) * TT + t];
      v = v * EM_ * (1.0f - s) + c;
      s = (v > 1.0f) ? 1.0f : 0.0f;
      y = A1_ * y1 + A2_ * y2 + s;
      y2 = y1; y1 = y;
    }
    psp[((size_t)b * TPAD + t) * FHIDP + h] = f2bf(y);
  }
}

// ---------------- GEMM3 (output layer current, M=10 in one 16-row tile) ---------
__global__ void gemm3_kernel(const __bf16* __restrict__ W3b,   // [16][512]
                             const __bf16* __restrict__ P,     // [B][TPAD][512]
                             const float*  __restrict__ b3,
                             float*        __restrict__ cur) { // [B][FOUT][TT]
  const int b    = blockIdx.x;
  const int t0   = blockIdx.y * 16;
  const int lane = threadIdx.x & 31;
  const int row  = lane & 15;
  const int half = lane >> 4;

  const __bf16* Ap = W3b + (size_t)row * FHIDP;
  const __bf16* Bp = P   + ((size_t)b * TPAD + (t0 + row)) * FHIDP;

  v8f acc = {};
  for (int k0 = 0; k0 < FHIDP; k0 += 32) {
    const int ka = k0 + 8 * half;
    const int kb = k0 + 8 * half;
    v16bf a  = load_frag(Ap + ka, Ap + ka + 16);
    v16bf bm = load_frag(Bp + kb, Bp + kb + 16);
    acc = __builtin_amdgcn_wmma_f32_16x16x32_bf16(false, a, false, bm, (short)0, acc, false, false);
  }

  const int t = t0 + row;
#pragma unroll
  for (int r = 0; r < 8; ++r) {
    int o = r + 8 * half;
    if (o < FOUT && t < TT)
      cur[((size_t)b * FOUT + o) * TT + t] = acc[r] + b3[o];
  }
}

// ---------------- final LIF scan -> spikes to d_out ----------------
__global__ void lif_out_kernel(const float* __restrict__ cur, float* __restrict__ out) {
  int idx = blockIdx.x * blockDim.x + threadIdx.x;
  if (idx >= BB * FOUT) return;
  int b = idx / FOUT, o = idx % FOUT;
  float v = 0.0f, s = 0.0f;
  for (int t = 0; t < TT; ++t) {
    float c = cur[((size_t)b * FOUT + o) * TT + t];
    v = v * EM_ * (1.0f - s) + c;
    s = (v > 1.0f) ? 1.0f : 0.0f;
    out[((size_t)b * FOUT + o) * TT + t] = s;
  }
}

// ---------------- host launch ----------------
extern "C" void kernel_launch(void* const* d_in, const int* in_sizes, int n_in,
                              void* d_out, int out_size, void* d_ws, size_t ws_size,
                              hipStream_t stream) {
  const float* x  = (const float*)d_in[0];  // [256,784,100]
  const float* W1 = (const float*)d_in[1];  // [1000,784]
  const float* b1 = (const float*)d_in[2];  // [1000]
  const float* W2 = (const float*)d_in[3];  // [500,500]
  const float* b2 = (const float*)d_in[4];  // [500]
  const float* W3 = (const float*)d_in[5];  // [10,500]
  const float* b3 = (const float*)d_in[6];  // [10]
  float* out = (float*)d_out;

  char* ws = (char*)d_ws;
  size_t off = 0;
  auto carve = [&](size_t bytes) { size_t o = off; off = (off + bytes + 255) & ~(size_t)255; return o; };

  __bf16* xb    = (__bf16*)(ws + carve((size_t)BB * TPAD * FINP * 2));
  __bf16* w1b   = (__bf16*)(ws + carve((size_t)1024 * FINP * 2));
  __bf16* w2b   = (__bf16*)(ws + carve((size_t)FHIDP * FHIDP * 2));
  __bf16* w3b   = (__bf16*)(ws + carve((size_t)16 * FHIDP * 2));
  float*  fbuf  = (float*)(ws + carve((size_t)BB * FHID * TT * 4));   // coding / cur2 / cur3 (serially reused)
  __bf16* psp2  = (__bf16*)(ws + carve((size_t)BB * TPAD * FHIDP * 2));
  __bf16* psp3  = (__bf16*)(ws + carve((size_t)BB * TPAD * FHIDP * 2));

  // 1) bf16 conversion + padding
  {
    size_t n = (size_t)BB * TPAD * FINP;
    cvt_x_kernel<<<dim3((unsigned)((n + 255) / 256)), 256, 0, stream>>>(x, xb);
  }
  cvt_pad_kernel<<<dim3((1024u * FINP + 255) / 256), 256, 0, stream>>>(W1, w1b, 1024, FINP, 2 * FHID, FIN);
  cvt_pad_kernel<<<dim3((FHIDP * FHIDP + 255) / 256), 256, 0, stream>>>(W2, w2b, FHIDP, FHIDP, FHID, FHID);
  cvt_pad_kernel<<<dim3((16u * FHIDP + 255) / 256), 256, 0, stream>>>(W3, w3b, 16, FHIDP, FOUT, FHID);

  // 2) GEMM1 + relu + sampling  -> coding (fbuf)
  gemm1_sample_kernel<<<dim3(BB, TPAD / 16, 8), 128, 0, stream>>>(w1b, xb, b1, fbuf);

  // 3) IIR -> psp2 (bf16, [b][t][h])
  iir_kernel<<<dim3((BB * FHIDP) / 256), 256, 0, stream>>>(fbuf, psp2);

  // 4) GEMM2 + bias -> cur2 (fbuf, reuse)
  gemm2_kernel<<<dim3(BB, TPAD / 16, 8), 128, 0, stream>>>(w2b, psp2, b2, fbuf);

  // 5) LIF + IIR -> psp3
  lif_iir_kernel<<<dim3((BB * FHIDP) / 256), 256, 0, stream>>>(fbuf, psp3);

  // 6) GEMM3 + bias -> cur3 (fbuf, reuse)
  gemm3_kernel<<<dim3(BB, TPAD / 16), 32, 0, stream>>>(w3b, psp3, b3, fbuf);

  // 7) final LIF -> output spikes
  lif_out_kernel<<<dim3((BB * FOUT + 255) / 256), 256, 0, stream>>>(fbuf, out);
}
